// RoutedExpertsNoEPGroupedMM_16226386444695
// MI455X (gfx1250) — matrix operand order repaired and checked by reading, not compiled, not run on hardware
//
#include <hip/hip_runtime.h>

// ---------------- problem constants (from reference) ----------------
constexpr int N  = 4096;
constexpr int K  = 4;
constexpr int E  = 32;
constexpr int D  = 2048;
constexpr int DI = 1408;
constexpr int NK = N * K;

// ---------------- tiling ----------------
constexpr int BM  = 128;   // rows per tile (token slots)
constexpr int BN  = 64;    // output cols per tile
constexpr int BK  = 32;    // k-step (bf16 WMMA depth)
constexpr int LDA = 40;    // LDS row stride in bf16 elems (80B: 16B-aligned, bank-spread)
constexpr int MAX_TILES = 256;   // >= E-1 + ceil(NK/BM) = 159

// ---------------- workspace layout (bytes) ----------------
constexpr size_t OFF_COUNTS = 0;
constexpr size_t OFF_CURS   = 256;
constexpr size_t OFF_STARTS = 512;
constexpr size_t OFF_NTILES = 768;
constexpr size_t OFF_TILES  = 1024;                       // MAX_TILES * sizeof(int4)
constexpr size_t OFF_S2A    = 8192;                       // NK * 4
constexpr size_t OFF_A2S    = OFF_S2A + (size_t)NK * 4;   // NK * 4
constexpr size_t OFF_H      = OFF_A2S + (size_t)NK * 4;   // NK * DI * 2 (bf16)
constexpr size_t OFF_ZO     = OFF_H + (size_t)NK * DI * 2;// NK * D * 2 (bf16)

// ---------------- WMMA types ----------------
typedef __attribute__((ext_vector_type(16))) __bf16 v16bf;
typedef __attribute__((ext_vector_type(8)))  float  v8f;

union FragU { v16bf v; uint4 u[2]; };

__device__ __forceinline__ unsigned short f2bf(float f) {
    unsigned u = __float_as_uint(f);
    u += 0x7FFFu + ((u >> 16) & 1u);       // round-to-nearest-even
    return (unsigned short)(u >> 16);
}
__device__ __forceinline__ float bf2f(unsigned short s) {
    return __uint_as_float((unsigned)s << 16);
}

__device__ __forceinline__ v8f bfwmma(v16bf a, v16bf b, v8f c) {
    return __builtin_amdgcn_wmma_f32_16x16x32_bf16(
        /*neg_a=*/false, a, /*neg_b=*/false, b,
        /*c_mod=*/(short)0, c, /*reuse_a=*/false, /*reuse_b=*/false);
}

// A fragment (16x32, 16-bit): lanes 0-15 row=lane hold K 0..7 (v0-3) and 16..23 (v4-7);
// lanes 16-31 row=lane-16 hold K 8..15 and 24..31.  (ISA 7.12.2)
__device__ __forceinline__ v16bf load_a_frag(const unsigned short* lds, int r0, int lane) {
    int row = r0 + (lane & 15);
    int k0  = (lane & 16) ? 8 : 0;
    const unsigned short* p = lds + row * LDA + k0;
    FragU f;
    f.u[0] = *(const uint4*)(p);       // K = k0 .. k0+7
    f.u[1] = *(const uint4*)(p + 16);  // K = k0+16 .. k0+23
    return f.v;
}
// B fragment (32x16, 16-bit): lanes 0-15 col=lane hold K 0..7 and 16..23; lanes 16-31 hold K 8..15 and 24..31.
__device__ __forceinline__ v16bf load_b_frag(const unsigned short* lds, int c0, int lane) {
    int col = c0 + (lane & 15);
    int k0  = (lane & 16) ? 8 : 0;
    const unsigned short* p = lds + col * LDA + k0;
    FragU f;
    f.u[0] = *(const uint4*)(p);       // K = k0 .. k0+7
    f.u[1] = *(const uint4*)(p + 16);  // K = k0+16 .. k0+23
    return f.v;
}

// async memory -> LDS copy of 16 bytes (tracked by ASYNCcnt)
__device__ __forceinline__ void async_copy_b128(unsigned lds_off, const void* gaddr) {
    asm volatile("global_load_async_to_lds_b128 %0, %1, off"
                 :: "v"(lds_off), "v"(gaddr) : "memory");
}
__device__ __forceinline__ void wait_asynccnt0() {
    asm volatile("s_wait_asynccnt 0x0" ::: "memory");
}

// ---------------- routing setup ----------------
__global__ void init_kernel(int* counts, int* cursors) {
    int t = threadIdx.x;
    if (t < E) { counts[t] = 0; cursors[t] = 0; }
}

__global__ void count_kernel(const int* __restrict__ indices, int* __restrict__ counts) {
    int a = blockIdx.x * blockDim.x + threadIdx.x;
    if (a < NK) atomicAdd(&counts[indices[a]], 1);
}

__global__ void plan_kernel(const int* __restrict__ counts, int* __restrict__ starts,
                            int* __restrict__ ntiles, int4* __restrict__ tiles) {
    if (threadIdx.x == 0 && blockIdx.x == 0) {
        int s = 0, nt = 0;
        for (int e = 0; e < E; ++e) {
            int c = counts[e];
            starts[e] = s;
            for (int t = 0; t < c; t += BM) {
                int r = c - t; if (r > BM) r = BM;
                tiles[nt++] = make_int4(e, s + t, r, 0);
            }
            s += c;
        }
        ntiles[0] = nt;
    }
}

__global__ void scatter_kernel(const int* __restrict__ indices, const int* __restrict__ starts,
                               int* __restrict__ cursors, int* __restrict__ s2a,
                               int* __restrict__ a2s) {
    int a = blockIdx.x * blockDim.x + threadIdx.x;
    if (a >= NK) return;
    int e = indices[a];
    int pos = starts[e] + atomicAdd(&cursors[e], 1);
    s2a[pos] = a;     // slot -> assignment id (token = a >> 2)
    a2s[a]   = pos;   // assignment -> slot
}

// ---------------- GEMM1: h = glu(x_gathered @ fc1^T), bf16 WMMA ----------------
__launch_bounds__(256)
__global__ void gemm1_kernel(const float* __restrict__ x, const float* __restrict__ w1,
                             const int* __restrict__ s2a, const int4* __restrict__ tiles,
                             const int* __restrict__ ntiles, unsigned short* __restrict__ h) {
    if ((int)blockIdx.x >= ntiles[0]) return;
    int4 t = tiles[blockIdx.x];
    const int e = t.x, slot0 = t.y, rows = t.z;
    const int col0 = blockIdx.y * BN;                 // y-column base in [0, DI)
    const int tid = threadIdx.x, lane = tid & 31, wave = tid >> 5;
    const int wr = wave & 3, wc = wave >> 2;          // 4x2 wave grid: 32 rows x 32 cols each

    __shared__ __align__(16) unsigned short lA [BM * LDA];
    __shared__ __align__(16) unsigned short lBy[BN * LDA];
    __shared__ __align__(16) unsigned short lBg[BN * LDA];
    __shared__ int tok_s[BM];

    // invalid rows clamp to token 0 (always-valid memory); their outputs are never stored
    if (tid < BM) tok_s[tid] = (tid < rows) ? (s2a[slot0 + tid] >> 2) : 0;
    __syncthreads();

    const float* w1y = w1 + ((size_t)e * (2 * DI) + col0) * D;
    const float* w1g = w1 + ((size_t)e * (2 * DI) + DI + col0) * D;

    // ---- hoist k-invariant row pointers out of the k-loop (no predication in loop) ----
    const int lrow = tid >> 3;           // 0..31
    const int lcol = (tid & 7) * 4;      // 0..28
    const float* xrow[4];
    unsigned short* aDst[4];
    #pragma unroll
    for (int p = 0; p < 4; ++p) {
        int r = p * 32 + lrow;
        xrow[p] = x + (size_t)tok_s[r] * D + lcol;
        aDst[p] = &lA[r * LDA + lcol];
    }
    const float* byp[2]; const float* bgp[2];
    unsigned short* byDst[2]; unsigned short* bgDst[2];
    #pragma unroll
    for (int p = 0; p < 2; ++p) {
        int r = p * 32 + lrow;
        byp[p] = w1y + (size_t)r * D + lcol;
        bgp[p] = w1g + (size_t)r * D + lcol;
        byDst[p] = &lBy[r * LDA + lcol];
        bgDst[p] = &lBg[r * LDA + lcol];
    }

    v8f accY[2][2], accG[2][2];
    v8f zero8 = {0.f, 0.f, 0.f, 0.f, 0.f, 0.f, 0.f, 0.f};
    for (int i = 0; i < 2; ++i)
        for (int j = 0; j < 2; ++j) { accY[i][j] = zero8; accG[i][j] = zero8; }

    for (int k0 = 0; k0 < D; k0 += BK) {
        // --- stage A tile (gathered x rows), fp32 -> bf16, unconditional ---
        #pragma unroll
        for (int p = 0; p < 4; ++p) {
            float4 v = *(const float4*)(xrow[p] + k0);
            unsigned short* dp = aDst[p];
            dp[0] = f2bf(v.x); dp[1] = f2bf(v.y); dp[2] = f2bf(v.z); dp[3] = f2bf(v.w);
        }
        // --- stage B tiles (y rows + gate rows of fc1), fp32 -> bf16 ---
        #pragma unroll
        for (int p = 0; p < 2; ++p) {
            float4 vy = *(const float4*)(byp[p] + k0);
            float4 vg = *(const float4*)(bgp[p] + k0);
            if (p == 0 && k0 + BK < D) {               // stream-ahead hint -> global_prefetch_b8
                __builtin_prefetch(byp[p] + k0 + BK, 0, 0);
                __builtin_prefetch(bgp[p] + k0 + BK, 0, 0);
            }
            unsigned short* dy = byDst[p];
            unsigned short* dg = bgDst[p];
            dy[0] = f2bf(vy.x); dy[1] = f2bf(vy.y); dy[2] = f2bf(vy.z); dy[3] = f2bf(vy.w);
            dg[0] = f2bf(vg.x); dg[1] = f2bf(vg.y); dg[2] = f2bf(vg.z); dg[3] = f2bf(vg.w);
        }
        __syncthreads();

        v16bf a0  = load_a_frag(lA, wr * 32,      lane);
        v16bf a1  = load_a_frag(lA, wr * 32 + 16, lane);
        v16bf by0 = load_b_frag(lBy, wc * 32,      lane);
        v16bf by1 = load_b_frag(lBy, wc * 32 + 16, lane);
        v16bf bg0 = load_b_frag(lBg, wc * 32,      lane);
        v16bf bg1 = load_b_frag(lBg, wc * 32 + 16, lane);

        accY[0][0] = bfwmma(a0, by0, accY[0][0]);
        accY[0][1] = bfwmma(a0, by1, accY[0][1]);
        accY[1][0] = bfwmma(a1, by0, accY[1][0]);
        accY[1][1] = bfwmma(a1, by1, accY[1][1]);
        accG[0][0] = bfwmma(a0, bg0, accG[0][0]);
        accG[0][1] = bfwmma(a0, bg1, accG[0][1]);
        accG[1][0] = bfwmma(a1, bg0, accG[1][0]);
        accG[1][1] = bfwmma(a1, bg1, accG[1][1]);
        __syncthreads();
    }

    // --- epilogue: h = y * silu(gate), bf16 store ---
    #pragma unroll
    for (int i = 0; i < 2; ++i)
        #pragma unroll
        for (int j = 0; j < 2; ++j)
            #pragma unroll
            for (int r = 0; r < 8; ++r) {
                int row = wr * 32 + i * 16 + r + ((lane >> 4) << 3);
                int col = col0 + wc * 32 + j * 16 + (lane & 15);
                if (row < rows) {
                    float yv = accY[i][j][r];
                    float gv = accG[i][j][r];
                    float hv = yv * (gv / (1.0f + __expf(-gv)));
                    h[(size_t)(slot0 + row) * DI + col] = f2bf(hv);
                }
            }
}

// ---------------- GEMM2: zo = h @ fc2^T (per expert), bf16 WMMA ----------------
__launch_bounds__(256)
__global__ void gemm2_kernel(const unsigned short* __restrict__ h, const float* __restrict__ w2,
                             const int4* __restrict__ tiles, const int* __restrict__ ntiles,
                             unsigned short* __restrict__ zo) {
    if ((int)blockIdx.x >= ntiles[0]) return;
    int4 t = tiles[blockIdx.x];
    const int e = t.x, slot0 = t.y, rows = t.z;
    const int col0 = blockIdx.y * BN;                 // d-column base in [0, D)
    const int tid = threadIdx.x, lane = tid & 31, wave = tid >> 5;
    const int wr = wave & 3, wc = wave >> 2;

    __shared__ __align__(16) unsigned short lA[BM * LDA];
    __shared__ __align__(16) unsigned short lB[BN * LDA];

    const float* w2e = w2 + ((size_t)e * D + col0) * DI;

    // ---- hoisted A-tile async copy descriptors (h is already bf16: mem->LDS direct) ----
    // invalid rows clamp to the tile's first row (always valid); their outputs are never stored
    unsigned aLds[2]; const unsigned short* aSrc[2];
    {
        const int arow = tid >> 2;        // 0..63
        const int acol = (tid & 3) * 8;   // 0,8,16,24 (bf16 elems; 16B chunks)
        #pragma unroll
        for (int p = 0; p < 2; ++p) {
            int r = p * 64 + arow;
            int rc = (r < rows) ? r : 0;
            aLds[p] = (unsigned)(size_t)&lA[r * LDA + acol];   // low 32 bits = LDS offset
            aSrc[p] = h + (size_t)(slot0 + rc) * DI + acol;
        }
    }
    const int lrow = tid >> 3;
    const int lcol = (tid & 7) * 4;
    const float* bp[2]; unsigned short* bDst[2];
    #pragma unroll
    for (int p = 0; p < 2; ++p) {
        int r = p * 32 + lrow;
        bp[p]   = w2e + (size_t)r * DI + lcol;
        bDst[p] = &lB[r * LDA + lcol];
    }

    v8f acc[2][2];
    v8f zero8 = {0.f, 0.f, 0.f, 0.f, 0.f, 0.f, 0.f, 0.f};
    for (int i = 0; i < 2; ++i)
        for (int j = 0; j < 2; ++j) acc[i][j] = zero8;

    for (int k0 = 0; k0 < DI; k0 += BK) {
        // --- stage A tile: async copy 128x32 bf16 straight into LDS (ASYNCcnt) ---
        #pragma unroll
        for (int p = 0; p < 2; ++p)
            async_copy_b128(aLds[p], (const void*)(aSrc[p] + k0));

        // --- stage B tile (fc2 rows), fp32 -> bf16: 64 rows x 32 ---
        #pragma unroll
        for (int p = 0; p < 2; ++p) {
            float4 v = *(const float4*)(bp[p] + k0);
            if (p == 0 && k0 + BK < DI) __builtin_prefetch(bp[p] + k0 + BK, 0, 0);
            unsigned short* dp = bDst[p];
            dp[0] = f2bf(v.x); dp[1] = f2bf(v.y); dp[2] = f2bf(v.z); dp[3] = f2bf(v.w);
        }
        wait_asynccnt0();
        __syncthreads();

        v16bf a0 = load_a_frag(lA, wr * 32,      lane);
        v16bf a1 = load_a_frag(lA, wr * 32 + 16, lane);
        v16bf b0 = load_b_frag(lB, wc * 32,      lane);
        v16bf b1 = load_b_frag(lB, wc * 32 + 16, lane);

        acc[0][0] = bfwmma(a0, b0, acc[0][0]);
        acc[0][1] = bfwmma(a0, b1, acc[0][1]);
        acc[1][0] = bfwmma(a1, b0, acc[1][0]);
        acc[1][1] = bfwmma(a1, b1, acc[1][1]);
        __syncthreads();
    }

    #pragma unroll
    for (int i = 0; i < 2; ++i)
        #pragma unroll
        for (int j = 0; j < 2; ++j)
            #pragma unroll
            for (int r = 0; r < 8; ++r) {
                int row = wr * 32 + i * 16 + r + ((lane >> 4) << 3);
                int col = col0 + wc * 32 + j * 16 + (lane & 15);
                if (row < rows)
                    zo[(size_t)(slot0 + row) * D + col] = f2bf(acc[i][j][r]);
            }
}

// ---------------- combine: out[n] = sum_k w[n,k] * zo[slot(n,k)] ----------------
__launch_bounds__(256)
__global__ void combine_kernel(const float* __restrict__ wts, const unsigned short* __restrict__ zo,
                               const int* __restrict__ a2s, float* __restrict__ out) {
    int g  = blockIdx.x * 256 + threadIdx.x;      // over N * (D/4)
    int n  = g >> 9;                              // D/4 = 512
    int dq = (g & 511) << 2;
    float4 wv = ((const float4*)wts)[n];
    float acc0 = 0.f, acc1 = 0.f, acc2 = 0.f, acc3 = 0.f;
    #pragma unroll
    for (int k = 0; k < 4; ++k) {
        int slot = a2s[n * 4 + k];
        float wk = (k == 0) ? wv.x : (k == 1) ? wv.y : (k == 2) ? wv.z : wv.w;
        ushort4 zv = *(const ushort4*)(zo + (size_t)slot * D + dq);
        acc0 += wk * bf2f(zv.x);
        acc1 += wk * bf2f(zv.y);
        acc2 += wk * bf2f(zv.z);
        acc3 += wk * bf2f(zv.w);
    }
    *(float4*)(out + (size_t)n * D + dq) = make_float4(acc0, acc1, acc2, acc3);
}

// ---------------- host launcher ----------------
extern "C" void kernel_launch(void* const* d_in, const int* in_sizes, int n_in,
                              void* d_out, int out_size, void* d_ws, size_t ws_size,
                              hipStream_t stream) {
    const float* x   = (const float*)d_in[0];   // [N, D]
    const float* wts = (const float*)d_in[1];   // [N, K]
    const float* w1  = (const float*)d_in[2];   // [E, 2*DI, D]
    const float* w2  = (const float*)d_in[3];   // [E, D, DI]
    const int*   idx = (const int*)d_in[4];     // [N, K]
    float* out = (float*)d_out;                 // [N, D]

    char* ws = (char*)d_ws;
    int*  counts  = (int*)(ws + OFF_COUNTS);
    int*  cursors = (int*)(ws + OFF_CURS);
    int*  starts  = (int*)(ws + OFF_STARTS);
    int*  ntiles  = (int*)(ws + OFF_NTILES);
    int4* tiles   = (int4*)(ws + OFF_TILES);
    int*  s2a     = (int*)(ws + OFF_S2A);
    int*  a2s     = (int*)(ws + OFF_A2S);
    unsigned short* hbuf = (unsigned short*)(ws + OFF_H);
    unsigned short* zo   = (unsigned short*)(ws + OFF_ZO);

    init_kernel<<<1, 64, 0, stream>>>(counts, cursors);
    count_kernel<<<NK / 256, 256, 0, stream>>>(idx, counts);
    plan_kernel<<<1, 32, 0, stream>>>(counts, starts, ntiles, tiles);
    scatter_kernel<<<NK / 256, 256, 0, stream>>>(idx, starts, cursors, s2a, a2s);

    gemm1_kernel<<<dim3(MAX_TILES, DI / BN), 256, 0, stream>>>(x, w1, s2a, tiles, ntiles, hbuf);
    gemm2_kernel<<<dim3(MAX_TILES, D / BN), 256, 0, stream>>>(hbuf, w2, tiles, ntiles, zo);

    combine_kernel<<<(N * (D / 4)) / 256, 256, 0, stream>>>(wts, zo, a2s, out);
}
